// Attention_18176301596931
// MI455X (gfx1250) — matrix operand order repaired and checked by reading, not compiled, not run on hardware
//
#include <hip/hip_runtime.h>
#include <hip/hip_bf16.h>

// ---------------------------------------------------------------------------
// Problem constants (B=1, S=2048, D=2048, NH=32, NKV=8, HD=64)
// ---------------------------------------------------------------------------
#define SEQ    2048
#define DMODEL 2048
#define NHEAD  32
#define NKVH   8
#define HDIM   64
#define EQKV   3072           // NH*HD + 2*NKV*HD
#define KOFF   2048           // start of k in qkv row
#define VOFF   2560           // start of v in qkv row

typedef __bf16 bf16;
typedef __attribute__((ext_vector_type(8)))  bf16  bf16x8;
typedef __attribute__((ext_vector_type(16))) bf16  v16bf;
typedef __attribute__((ext_vector_type(8)))  float v8f;

#define WMMA_BF16(a, b, c) \
  __builtin_amdgcn_wmma_f32_16x16x32_bf16(false, (a), false, (b), (short)0, (c), false, false)

// ---------------------------------------------------------------------------
// Fragment loaders (documented CDNA5 16-bit WMMA layouts, wave32)
//  A (16x32, MxK): lane(lm,half) holds row lm, K = {half*8+0..7, 16+half*8+0..7}
//  B (32x16, KxN): lane(lm,half) holds col lm, K = {half*8+0..7, 16+half*8+0..7}
// ---------------------------------------------------------------------------
__device__ __forceinline__ v16bf load_frag_a(const bf16* p, int ld, int row0,
                                             int k0, int lm, int half) {
  const bf16* base = p + (size_t)(row0 + lm) * (size_t)ld + k0 + half * 8;
  bf16x8 g0 = *(const bf16x8*)(base);
  bf16x8 g1 = *(const bf16x8*)(base + 16);
  v16bf r;
#pragma unroll
  for (int i = 0; i < 8; ++i) { r[i] = g0[i]; r[i + 8] = g1[i]; }
  return r;
}

__device__ __forceinline__ v16bf load_frag_b(const bf16* p, int ld, int col0,
                                             int k0, int lm, int half) {
  const bf16* base = p + (size_t)(col0 + lm) * (size_t)ld + k0 + half * 8;
  bf16x8 g0 = *(const bf16x8*)(base);
  bf16x8 g1 = *(const bf16x8*)(base + 16);
  v16bf r;
#pragma unroll
  for (int i = 0; i < 8; ++i) { r[i] = g0[i]; r[i + 8] = g1[i]; }
  return r;
}

// ---------------------------------------------------------------------------
// fp32 -> bf16 cast
// ---------------------------------------------------------------------------
__global__ __launch_bounds__(256) void cast_f32_bf16(const float* __restrict__ in,
                                                     bf16* __restrict__ out, int n) {
  int i = blockIdx.x * blockDim.x + threadIdx.x;
  if (i < n) out[i] = (bf16)in[i];
}

// ---------------------------------------------------------------------------
// GEMM: C[M,N] (f32) = A[M,K](bf16, row-major) * B[N,K](bf16, row-major)^T
// block = 256 threads = 8 waves (2 row x 4 col), wave tile = 64x64 (4x4 WMMA)
// grid  = (N/256, M/128)
// 16 WMMAs per 16 b128-loads per 32-deep K step -> ~32 FLOP/B from L2.
// ---------------------------------------------------------------------------
__global__ __launch_bounds__(256) void gemm_abt_bf16(const bf16* __restrict__ A,
                                                     const bf16* __restrict__ B,
                                                     float* __restrict__ C,
                                                     int M, int N, int K) {
  const int lane = threadIdx.x & 31;
  const int w    = threadIdx.x >> 5;
  const int half = lane >> 4;
  const int lm   = lane & 15;
  const int row0 = blockIdx.y * 128 + (w & 1) * 64;
  const int col0 = blockIdx.x * 256 + (w >> 1) * 64;

  v8f acc[4][4];
#pragma unroll
  for (int i = 0; i < 4; ++i)
#pragma unroll
    for (int j = 0; j < 4; ++j) acc[i][j] = (v8f){};

  for (int k0 = 0; k0 < K; k0 += 32) {
    if (k0 + 64 < K) {   // lowers to global_prefetch_b8
      __builtin_prefetch(A + (size_t)(row0 + lm) * K + k0 + 64, 0, 1);
      __builtin_prefetch(B + (size_t)(col0 + lm) * K + k0 + 64, 0, 1);
    }
    v16bf fa[4], fb[4];
#pragma unroll
    for (int i = 0; i < 4; ++i) fa[i] = load_frag_a(A, K, row0 + i * 16, k0, lm, half);
#pragma unroll
    for (int j = 0; j < 4; ++j) fb[j] = load_frag_b(B, K, col0 + j * 16, k0, lm, half);
#pragma unroll
    for (int i = 0; i < 4; ++i)
#pragma unroll
      for (int j = 0; j < 4; ++j) acc[i][j] = WMMA_BF16(fa[i], fb[j], acc[i][j]);
  }

  // C/D layout: VGPR r -> row half*8+r (relative to 16-row subtile), col = lm
#pragma unroll
  for (int i = 0; i < 4; ++i)
#pragma unroll
    for (int j = 0; j < 4; ++j)
#pragma unroll
      for (int r = 0; r < 8; ++r)
        C[(size_t)(row0 + i * 16 + half * 8 + r) * N + col0 + j * 16 + lm] = acc[i][j][r];
}

// ---------------------------------------------------------------------------
// RoPE + repack: q -> qbf[NH][S][HD], k -> kbf[NKV][S][HD], v -> vT[NKV][HD][S]
// ---------------------------------------------------------------------------
__global__ __launch_bounds__(256) void rope_pack_q(const float* __restrict__ qkv,
                                                   const float* __restrict__ cosT,
                                                   const float* __restrict__ sinT,
                                                   bf16* __restrict__ qbf) {
  int idx = blockIdx.x * blockDim.x + threadIdx.x;     // SEQ*NHEAD*32 pairs
  if (idx >= SEQ * NHEAD * 32) return;
  int pr = idx & 31, h = (idx >> 5) & 31, s = idx >> 10;
  const float* base = qkv + (size_t)s * EQKV + h * HDIM + 2 * pr;
  float tr = base[0], ti = base[1];
  float c = cosT[s * 32 + pr], sn = sinT[s * 32 + pr];
  bf16* o = qbf + ((size_t)h * SEQ + s) * HDIM + 2 * pr;
  o[0] = (bf16)(tr * c - ti * sn);
  o[1] = (bf16)(tr * sn + ti * c);
}

__global__ __launch_bounds__(256) void rope_pack_k(const float* __restrict__ qkv,
                                                   const float* __restrict__ cosT,
                                                   const float* __restrict__ sinT,
                                                   bf16* __restrict__ kbf) {
  int idx = blockIdx.x * blockDim.x + threadIdx.x;     // SEQ*NKVH*32 pairs
  if (idx >= SEQ * NKVH * 32) return;
  int pr = idx & 31, h = (idx >> 5) & 7, s = idx >> 8;
  const float* base = qkv + (size_t)s * EQKV + KOFF + h * HDIM + 2 * pr;
  float tr = base[0], ti = base[1];
  float c = cosT[s * 32 + pr], sn = sinT[s * 32 + pr];
  bf16* o = kbf + ((size_t)h * SEQ + s) * HDIM + 2 * pr;
  o[0] = (bf16)(tr * c - ti * sn);
  o[1] = (bf16)(tr * sn + ti * c);
}

__global__ __launch_bounds__(256) void pack_v_t(const float* __restrict__ qkv,
                                                bf16* __restrict__ vT) {
  int idx = blockIdx.x * blockDim.x + threadIdx.x;     // SEQ*NKVH*HDIM
  if (idx >= SEQ * NKVH * HDIM) return;
  int e = idx & 63, h = (idx >> 6) & 7, s = idx >> 9;
  float v = qkv[(size_t)s * EQKV + VOFF + h * HDIM + e];
  vT[((size_t)h * HDIM + e) * SEQ + s] = (bf16)v;       // [NKV][HD][S]
}

// ---------------------------------------------------------------------------
// Causal flash attention. One wave32 per (head, 32-row q block = 2 tiles).
// block = 128 threads (4 waves); grid = NH*(S/32)/4 = 512 blocks.
// K/V fragments are loaded once per 32-key block and reused by both q tiles.
// Online softmax; P relayout via per-wave LDS tile (ds_store -> s_wait_dscnt
// -> ds_load in A-layout). Output written bf16 into ybf[S][D] at col h*HD.
// ---------------------------------------------------------------------------
__global__ __launch_bounds__(128) void flash_attn(const bf16* __restrict__ qbf,
                                                  const bf16* __restrict__ kbf,
                                                  const bf16* __restrict__ vT,
                                                  bf16* __restrict__ ybf) {
  __shared__ __align__(16) bf16 plds[4][2][16 * 32];

  const int w    = threadIdx.x >> 5;
  const int lane = threadIdx.x & 31;
  const int half = lane >> 4;
  const int lm   = lane & 15;

  const int gw    = blockIdx.x * 4 + w;    // global wave id
  const int h     = gw >> 6;               // 64 q-blocks (32 rows) per head
  const int qblk  = gw & 63;
  const int qbase = qblk * 32;
  const int hkv   = h >> 2;                // GQA rep = 4

  const bf16* Q = qbf + (size_t)h   * SEQ * HDIM;
  const bf16* K = kbf + (size_t)hkv * SEQ * HDIM;
  const bf16* V = vT  + (size_t)hkv * HDIM * SEQ;

  // Q block 32x64 as 2 tiles x 2 A fragments (persist across all KV blocks)
  v16bf qa[2][2];
#pragma unroll
  for (int t = 0; t < 2; ++t) {
    qa[t][0] = load_frag_a(Q, HDIM, qbase + t * 16, 0,  lm, half);
    qa[t][1] = load_frag_a(Q, HDIM, qbase + t * 16, 32, lm, half);
  }

  const float NEG_INF = -__builtin_inff();
  float mrun[2][8], lrun[2][8];
  v8f o[2][4];
#pragma unroll
  for (int t = 0; t < 2; ++t) {
#pragma unroll
    for (int r = 0; r < 8; ++r) { mrun[t][r] = NEG_INF; lrun[t][r] = 0.f; }
#pragma unroll
    for (int cb = 0; cb < 4; ++cb) o[t][cb] = (v8f){};
  }

  const int npairs = (qbase + 63) >> 5;    // 32-key blocks covering keys <= qbase+31

  for (int p = 0; p < npairs; ++p) {
    const int kb = p * 32;

    // ---- K fragments, shared by both q tiles ----------------------------
    v16bf bk0a = load_frag_b(K, HDIM, kb,      0,  lm, half);
    v16bf bk0b = load_frag_b(K, HDIM, kb,      32, lm, half);
    v16bf bk1a = load_frag_b(K, HDIM, kb + 16, 0,  lm, half);
    v16bf bk1b = load_frag_b(K, HDIM, kb + 16, 32, lm, half);

    // ---- S = Q * K^T: 2 tiles x 2 score fragments (8 WMMAs) -------------
    v8f s[2][2];
#pragma unroll
    for (int t = 0; t < 2; ++t) {
      v8f s0 = {}, s1 = {};
      s0 = WMMA_BF16(qa[t][0], bk0a, s0);
      s0 = WMMA_BF16(qa[t][1], bk0b, s0);
      s1 = WMMA_BF16(qa[t][0], bk1a, s1);
      s1 = WMMA_BF16(qa[t][1], bk1b, s1);
      s[t][0] = s0; s[t][1] = s1;
    }

    // ---- scale + causal mask + online softmax ---------------------------
#pragma unroll
    for (int t = 0; t < 2; ++t) {
#pragma unroll
      for (int r = 0; r < 8; ++r) {
        const int qi = qbase + t * 16 + half * 8 + r;   // global query row
        float v0 = s[t][0][r] * 0.125f;                 // 1/sqrt(64)
        float v1 = s[t][1][r] * 0.125f;
        if (kb + lm > qi)      v0 = NEG_INF;
        if (kb + 16 + lm > qi) v1 = NEG_INF;

        // row max across the 16 lanes holding this row (bits 0..3 only)
        float mr = fmaxf(v0, v1);
        mr = fmaxf(mr, __shfl_xor(mr, 1));
        mr = fmaxf(mr, __shfl_xor(mr, 2));
        mr = fmaxf(mr, __shfl_xor(mr, 4));
        mr = fmaxf(mr, __shfl_xor(mr, 8));

        const float mnew = fmaxf(mrun[t][r], mr);
        const float corr = expf(mrun[t][r] - mnew);
        const float p0   = expf(v0 - mnew);
        const float p1   = expf(v1 - mnew);
        float rs = p0 + p1;
        rs += __shfl_xor(rs, 1);
        rs += __shfl_xor(rs, 2);
        rs += __shfl_xor(rs, 4);
        rs += __shfl_xor(rs, 8);

        lrun[t][r] = lrun[t][r] * corr + rs;
        mrun[t][r] = mnew;

        // stage P tile (16x32) in LDS, C-layout -> row-major
        plds[w][t][(half * 8 + r) * 32 + lm]      = (bf16)p0;
        plds[w][t][(half * 8 + r) * 32 + 16 + lm] = (bf16)p1;

#pragma unroll
        for (int cb = 0; cb < 4; ++cb) o[t][cb][r] *= corr;
      }
    }

    // all DS stores by this wave must land before re-reading in A-layout
    asm volatile("s_wait_dscnt 0" ::: "memory");

    // ---- O += P(16x32) * V(32x64); V fragments shared by both tiles -----
    v16bf pa0 = load_frag_a(&plds[w][0][0], 32, 0, 0, lm, half);
    v16bf pa1 = load_frag_a(&plds[w][1][0], 32, 0, 0, lm, half);
#pragma unroll
    for (int cb = 0; cb < 4; ++cb) {
      v16bf bv = load_frag_b(V, SEQ, cb * 16, kb, lm, half);  // vT is K-contig
      o[0][cb] = WMMA_BF16(pa0, bv, o[0][cb]);
      o[1][cb] = WMMA_BF16(pa1, bv, o[1][cb]);
    }
  }

  // ---- finalize: divide by row sums, emit bf16 into y[S][D] -------------
#pragma unroll
  for (int t = 0; t < 2; ++t)
#pragma unroll
    for (int r = 0; r < 8; ++r) {
      const int row  = qbase + t * 16 + half * 8 + r;
      const float il = 1.0f / lrun[t][r];
#pragma unroll
      for (int cb = 0; cb < 4; ++cb)
        ybf[(size_t)row * DMODEL + h * HDIM + cb * 16 + lm] = (bf16)(o[t][cb][r] * il);
    }
}

// ---------------------------------------------------------------------------
// Host launcher
// ---------------------------------------------------------------------------
extern "C" void kernel_launch(void* const* d_in, const int* in_sizes, int n_in,
                              void* d_out, int out_size, void* d_ws, size_t ws_size,
                              hipStream_t stream) {
  const float* x    = (const float*)d_in[0];   // [S][D]
  const float* cosT = (const float*)d_in[1];   // [S][32]
  const float* sinT = (const float*)d_in[2];   // [S][32]
  const float* wqkv = (const float*)d_in[3];   // [EQKV][D]
  const float* wo   = (const float*)d_in[4];   // [D][D]
  float* out = (float*)d_out;

  // workspace carve-up (bytes)
  char* ws = (char*)d_ws;
  size_t off = 0;
  auto carve = [&](size_t bytes) { char* p = ws + off; off += (bytes + 255) & ~(size_t)255; return p; };
  bf16*  xbf    = (bf16*) carve((size_t)SEQ * DMODEL * 2);
  bf16*  wqkvbf = (bf16*) carve((size_t)EQKV * DMODEL * 2);
  bf16*  wobf   = (bf16*) carve((size_t)DMODEL * DMODEL * 2);
  float* qkv    = (float*)carve((size_t)SEQ * EQKV * 4);
  bf16*  qbf    = (bf16*) carve((size_t)NHEAD * SEQ * HDIM * 2);
  bf16*  kbf    = (bf16*) carve((size_t)NKVH * SEQ * HDIM * 2);
  bf16*  vT     = (bf16*) carve((size_t)NKVH * HDIM * SEQ * 2);
  bf16*  ybf    = (bf16*) carve((size_t)SEQ * DMODEL * 2);

  // 1) casts to bf16
  {
    int n1 = SEQ * DMODEL, n2 = EQKV * DMODEL, n3 = DMODEL * DMODEL;
    cast_f32_bf16<<<(n1 + 255) / 256, 256, 0, stream>>>(x, xbf, n1);
    cast_f32_bf16<<<(n2 + 255) / 256, 256, 0, stream>>>(wqkv, wqkvbf, n2);
    cast_f32_bf16<<<(n3 + 255) / 256, 256, 0, stream>>>(wo, wobf, n3);
  }

  // 2) qkv = x * wqkv^T   (M=2048, N=3072, K=2048)
  gemm_abt_bf16<<<dim3(EQKV / 256, SEQ / 128), 256, 0, stream>>>(
      xbf, wqkvbf, qkv, SEQ, EQKV, DMODEL);

  // 3) RoPE + repack
  rope_pack_q<<<(SEQ * NHEAD * 32 + 255) / 256, 256, 0, stream>>>(qkv, cosT, sinT, qbf);
  rope_pack_k<<<(SEQ * NKVH * 32 + 255) / 256, 256, 0, stream>>>(qkv, cosT, sinT, kbf);
  pack_v_t   <<<(SEQ * NKVH * HDIM + 255) / 256, 256, 0, stream>>>(qkv, vT);

  // 4) causal flash attention (512 blocks x 4 waves, 32 q-rows per wave)
  flash_attn<<<(NHEAD * (SEQ / 32)) / 4, 128, 0, stream>>>(qbf, kbf, vT, ybf);

  // 5) out = y * wo^T   (M=2048, N=2048, K=2048)
  gemm_abt_bf16<<<dim3(DMODEL / 256, SEQ / 128), 256, 0, stream>>>(
      ybf, wobf, out, SEQ, DMODEL, DMODEL);

  (void)in_sizes; (void)n_in; (void)out_size; (void)ws_size;
}
